// Inferer_48498770706560
// MI455X (gfx1250) — compile-verified, matrix operand order checked by reading, not run on hardware
//
#include <hip/hip_runtime.h>

// ---------------------------------------------------------------------------
// Types for CDNA5 WMMA (wave32, 16x16x32 bf16 -> f32)
// ---------------------------------------------------------------------------
typedef __attribute__((ext_vector_type(16))) __bf16 bf16x16;
typedef __attribute__((ext_vector_type(8)))  __bf16 bf16x8;
typedef __attribute__((ext_vector_type(8)))  float  f32x8;
typedef __attribute__((ext_vector_type(4)))  float  f32x4;
typedef __attribute__((ext_vector_type(4)))  int    i32x4;

#define WMMA_BF16(a, b, c) \
    __builtin_amdgcn_wmma_f32_16x16x32_bf16(false, (a), false, (b), (short)0, (c), false, false)

// gfx1250 async global->LDS copy path (guarded: falls back to ld/st if the
// toolchain doesn't expose the builtins).
#if defined(__has_builtin)
#if __has_builtin(__builtin_amdgcn_global_load_async_to_lds_b128) && \
    __has_builtin(__builtin_amdgcn_s_wait_asynccnt)
#define HAVE_ASYNC_LDS 1
#endif
#endif

// ---------------------------------------------------------------------------
// Problem constants (match reference)
// ---------------------------------------------------------------------------
constexpr int Bn   = 256;
constexpr int NO_A = 64,  NF_A = 4;
constexpr int NO_B = 32,  NF_B = 3;
constexpr int NS   = 7;                  // NF_A + NF_B
constexpr int DIN  = 128, DL = 64, DG = 64, DH = 256;
constexpr int DGE  = 448;                // DG*NF_A + DG*NF_B
constexpr int DINdA = 704;               // DGE + DL*NF_A
constexpr int DINdB = 640;               // DGE + DL*NF_B
constexpr int NP   = 2;
constexpr int MA   = Bn * NO_A * NF_A;   // 65536 encoder rows (class A)
constexpr int MB   = Bn * NO_B * NF_B;   // 24576 encoder rows (class B)

__device__ inline f32x8 zero8() {
    f32x8 z = {0.f, 0.f, 0.f, 0.f, 0.f, 0.f, 0.f, 0.f};
    return z;
}

// Load one 16-half fragment: 8 contiguous halfs at p, 8 more at p+16.
// (A-matrix 16x32 bf16 layout: lanes 0-15 hold K{0..7,16..23}, lanes 16-31
//  hold K{8..15,24..31} — caller bakes the +8 lane offset into p.)
__device__ inline bf16x16 frag_from(const __bf16* p) {
    bf16x8 lo = *(const bf16x8*)p;
    bf16x8 hi = *(const bf16x8*)(p + 16);
    bf16x16 r;
#pragma unroll
    for (int i = 0; i < 8; ++i) { r[i] = lo[i]; r[i + 8] = hi[i]; }
    return r;
}

// Same fragment shape but sourced from fp32 memory, converting to bf16.
__device__ inline bf16x16 frag_from_f32(const float* p) {
    f32x4 a = *(const f32x4*)(p);
    f32x4 b = *(const f32x4*)(p + 4);
    f32x4 c = *(const f32x4*)(p + 16);
    f32x4 d = *(const f32x4*)(p + 20);
    bf16x16 r;
#pragma unroll
    for (int i = 0; i < 4; ++i) {
        r[i]      = (__bf16)a[i];
        r[4 + i]  = (__bf16)b[i];
        r[8 + i]  = (__bf16)c[i];
        r[12 + i] = (__bf16)d[i];
    }
    return r;
}

// ---------------------------------------------------------------------------
// Prep: pack encoder weights [Wl (128x64) || Wg (128x64)] -> bf16, N-major
// dst layout: [col 0..127][k 0..127]
// ---------------------------------------------------------------------------
__global__ void pack_encw_kernel(const float* __restrict__ Wl,
                                 const float* __restrict__ Wg,
                                 __bf16* __restrict__ dst) {
    int idx = blockIdx.x * blockDim.x + threadIdx.x;
    if (idx >= 128 * 128) return;
    int col = idx >> 7, k = idx & 127;
    float v = (col < 64) ? Wl[k * 64 + col] : Wg[k * 64 + (col - 64)];
    dst[col * 128 + k] = (__bf16)v;
}

// Prep: transpose (S,K,N) f32 -> (S,N,K) bf16
__global__ void transpose_bf16_kernel(const float* __restrict__ src,
                                      __bf16* __restrict__ dst,
                                      int S, int K, int N) {
    long total = (long)S * K * N;
    for (long idx = (long)blockIdx.x * blockDim.x + threadIdx.x; idx < total;
         idx += (long)gridDim.x * blockDim.x) {
        long s   = idx / ((long)K * N);
        long rem = idx - s * (long)K * N;
        int  k   = (int)(rem / N);
        int  n   = (int)(rem - (long)k * N);
        dst[(s * N + n) * (long)K + k] = (__bf16)src[idx];
    }
}

// ---------------------------------------------------------------------------
// Encoder GEMM: (M x 128) @ (128 x 128) with fused epilogue.
// cols 0..63  -> hl = relu(. + bl)    (bf16, decoder local features)
// cols 64..127-> hg = . + bg          (f32, pre-pool global features)
// Block: 256 threads = 8 waves; tile 64 rows x 128 cols; wave does 4 16x16.
// The 32KB weight matrix is staged once per block into LDS (async-to-LDS on
// toolchains that expose it), so all B-fragments come from ds_load_b128.
// ---------------------------------------------------------------------------
constexpr int WSTRIDE = 136;   // 128 + 8 halfs pad -> conflict-free ds_load_b128

__global__ void enc_gemm_kernel(const float* __restrict__ x,
                                const __bf16* __restrict__ Wt,   // (128 cols,128 k)
                                const float* __restrict__ bl,
                                const float* __restrict__ bg,
                                __bf16* __restrict__ hl,          // (M,64)
                                float*  __restrict__ hg) {        // (M,64)
    __shared__ __bf16 wlds[128 * WSTRIDE];   // 34.8 KB

    const int tid  = threadIdx.x;
    const int lane = tid & 31;
    const int wave = tid >> 5;             // 0..7
    const int m0   = blockIdx.x * 64;
    const int rt   = wave & 3;             // row tile 0..3
    const int cb   = wave >> 2;            // column interleave 0..1

    // ---- stage weights: 2048 chunks of 8 halfs (16B), 8 per thread
#pragma unroll
    for (int it = 0; it < 8; ++it) {
        int c   = tid + it * 256;
        int row = c >> 4;
        int off = (c & 15) * 8;
        const __bf16* src = Wt + row * 128 + off;
        __bf16*       dst = &wlds[row * WSTRIDE + off];
#if defined(HAVE_ASYNC_LDS)
        __builtin_amdgcn_global_load_async_to_lds_b128(
            (__attribute__((address_space(1))) i32x4*)src,
            (__attribute__((address_space(3))) i32x4*)dst, 0, 0);
#else
        *(bf16x8*)dst = *(const bf16x8*)src;
#endif
    }
#if defined(HAVE_ASYNC_LDS)
    __builtin_amdgcn_s_wait_asynccnt(0);
#endif
    __syncthreads();

    f32x8 acc[4];
#pragma unroll
    for (int i = 0; i < 4; ++i) acc[i] = zero8();

    const int arow = m0 + rt * 16 + (lane & 15);
    const int koff = (lane >> 4) * 8;

    __builtin_prefetch(x + (size_t)(arow + 64) * DIN, 0, 1);  // global_prefetch_b8

#pragma unroll
    for (int kt = 0; kt < 4; ++kt) {
        bf16x16 afrag = frag_from_f32(x + (size_t)arow * DIN + kt * 32 + koff);
#pragma unroll
        for (int i = 0; i < 4; ++i) {
            int ct = cb + i * 2;           // covers 0..7 across the 2 wave groups
            acc[i] = WMMA_BF16(
                afrag,
                frag_from(&wlds[(ct * 16 + (lane & 15)) * WSTRIDE + kt * 32 + koff]),
                acc[i]);
        }
    }

    // Epilogue.  C layout: col = lane&15, row = v + (lane>>4)*8.
    const int row0  = m0 + rt * 16 + (lane >> 4) * 8;
    const int col_l = lane & 15;
#pragma unroll
    for (int i = 0; i < 4; ++i) {
        int col = (cb + i * 2) * 16 + col_l;
#pragma unroll
        for (int v = 0; v < 8; ++v) {
            int   row = row0 + v;
            float val = acc[i][v];
            if (col < 64) {
                float o = val + bl[col];
                hl[(size_t)row * 64 + col] = (__bf16)(o > 0.f ? o : 0.f);
            } else {
                hg[(size_t)row * 64 + (col - 64)] = val + bg[col - 64];
            }
        }
    }
}

// ---------------------------------------------------------------------------
// Object-masked sum pool over objects + relu -> r (B,7,64)  (f32 + bf16)
// grid (B, 7), block 64 (one thread per DG channel)
// ---------------------------------------------------------------------------
__global__ void reduce_global_kernel(const float* __restrict__ hgA,
                                     const float* __restrict__ hgB,
                                     const int* __restrict__ objA,
                                     const int* __restrict__ objB,
                                     float*  __restrict__ r32,
                                     __bf16* __restrict__ rbf) {
    int b = blockIdx.x, fg = blockIdx.y, t = threadIdx.x;
    float acc = 0.f;
    if (fg < NF_A) {
        for (int o = 0; o < NO_A; ++o)
            if (objA[b * NO_A + o])
                acc += hgA[(size_t)(((b * NO_A + o) * NF_A) + fg) * 64 + t];
    } else {
        int f = fg - NF_A;
        for (int o = 0; o < NO_B; ++o)
            if (objB[b * NO_B + o])
                acc += hgB[(size_t)(((b * NO_B + o) * NF_B) + f) * 64 + t];
    }
    acc = acc > 0.f ? acc : 0.f;
    r32[(b * NS + fg) * 64 + t] = acc;
    rbf[(b * NS + fg) * 64 + t] = (__bf16)acc;
}

// ---------------------------------------------------------------------------
// Fused per-field decoder: assemble masked (local || global) input in LDS,
// then lrelu(x@W1+b1) -> lrelu(h@W2+b2) -> h@Wp+bp, fully inside one block.
// Block: 256 threads = 8 waves; tile = 32 rows x 256 cols per layer.
// grid: (M/32, NF)
// ---------------------------------------------------------------------------
__global__ void decoder_kernel(const __bf16* __restrict__ hl,     // (B*NO*NF,64)
                               const __bf16* __restrict__ rbf,    // (B,7,64)
                               const int*    __restrict__ lmask,  // (B,NF,NF)
                               const int*    __restrict__ gmask,  // (B,7,7)
                               const __bf16* __restrict__ W1t,    // (NF,256,DINd)
                               const float*  __restrict__ b1,     // (NF,256)
                               const __bf16* __restrict__ W2t,    // (NF,256,256)
                               const float*  __restrict__ b2,     // (NF,256)
                               const float*  __restrict__ Wp,     // (NF,256,2)
                               const float*  __restrict__ bp,     // (NF,2)
                               float* __restrict__ out,           // (B,NO,NF,2)
                               int NO, int NF, int DINd, int sglob_base) {
    __shared__ __bf16 smem[32 * (DINdA + 8)];   // 45.5 KB, reused across layers

    const int s     = blockIdx.y;
    const int m0    = blockIdx.x * 32;
    const int b     = m0 / NO;                  // 32 | NO -> single batch per block
    const int tid   = threadIdx.x;
    const int lane  = tid & 31;
    const int wave  = tid >> 5;
    const int LOCAL = NF * 64;
    const int STRIDE = DINd + 8;                // pad to dodge LDS bank conflicts

    // ---- assemble masked decoder input tile (32 x DINd) in LDS (bf16)
    for (int idx = tid; idx < 32 * DINd; idx += 256) {
        int rr = idx / DINd, d = idx - rr * DINd;
        int m = m0 + rr, o = m - b * NO;
        __bf16 v = (__bf16)0.0f;
        if (d < LOCAL) {
            int f = d >> 6, c = d & 63;
            if (lmask[(b * NF + s) * NF + f])
                v = hl[(size_t)((b * NO + o) * NF + f) * 64 + c];
        } else {
            int dg = d - LOCAL, fgl = dg >> 6, c = dg & 63;
            if (gmask[(b * NS + (sglob_base + s)) * NS + fgl])
                v = rbf[(b * NS + fgl) * 64 + c];
        }
        smem[rr * STRIDE + d] = v;
    }
    __syncthreads();

    const int rt   = wave & 1;                 // row tile 0..1
    const int cgrp = wave >> 1;                // column group 0..3
    const int rloc = rt * 16 + (lane & 15);
    const int koff = (lane >> 4) * 8;
    const int row0 = rt * 16 + (lane >> 4) * 8;
    const int col_l = lane & 15;

    // ---- layer 1: (32 x DINd) @ (DINd x 256)
    f32x8 acc[4];
#pragma unroll
    for (int i = 0; i < 4; ++i) acc[i] = zero8();
    const int KT1 = DINd / 32;
    for (int kt = 0; kt < KT1; ++kt) {
        bf16x16 afrag = frag_from(&smem[rloc * STRIDE + kt * 32 + koff]);
#pragma unroll
        for (int i = 0; i < 4; ++i) {
            int ct = cgrp + i * 4;             // 0..15
            const __bf16* bptr =
                W1t + (size_t)(s * 256 + ct * 16 + col_l) * DINd + kt * 32 + koff;
            acc[i] = WMMA_BF16(afrag, frag_from(bptr), acc[i]);
        }
    }
    __syncthreads();   // everyone done reading input region before overwrite

    // h1 -> LDS (stride 264), bias + lrelu, bf16
#pragma unroll
    for (int i = 0; i < 4; ++i) {
        int   col  = (cgrp + i * 4) * 16 + col_l;
        float bias = b1[s * 256 + col];
#pragma unroll
        for (int v = 0; v < 8; ++v) {
            float h = acc[i][v] + bias;
            h = h >= 0.f ? h : 0.01f * h;
            smem[(row0 + v) * 264 + col] = (__bf16)h;
        }
    }
    __syncthreads();

    // ---- layer 2: (32 x 256) @ (256 x 256)
#pragma unroll
    for (int i = 0; i < 4; ++i) acc[i] = zero8();
#pragma unroll
    for (int kt = 0; kt < 8; ++kt) {
        bf16x16 afrag = frag_from(&smem[rloc * 264 + kt * 32 + koff]);
#pragma unroll
        for (int i = 0; i < 4; ++i) {
            int ct = cgrp + i * 4;
            const __bf16* bptr =
                W2t + (size_t)(s * 256 + ct * 16 + col_l) * 256 + kt * 32 + koff;
            acc[i] = WMMA_BF16(afrag, frag_from(bptr), acc[i]);
        }
    }

    // h2 -> LDS at offset 8448 (disjoint from h1 region), bias + lrelu
    const int H2 = 32 * 264;   // 8448
#pragma unroll
    for (int i = 0; i < 4; ++i) {
        int   col  = (cgrp + i * 4) * 16 + col_l;
        float bias = b2[s * 256 + col];
#pragma unroll
        for (int v = 0; v < 8; ++v) {
            float h = acc[i][v] + bias;
            h = h >= 0.f ? h : 0.01f * h;
            smem[H2 + (row0 + v) * 264 + col] = (__bf16)h;
        }
    }
    __syncthreads();

    // ---- layer 3: (32 x 256) @ (256 x 2), plain VALU (N too small for WMMA)
    if (tid < 64) {
        int   rr = tid & 31, p = tid >> 5;
        float a3 = bp[s * NP + p];
        for (int e = 0; e < 256; ++e)
            a3 += (float)smem[H2 + rr * 264 + e] * Wp[(s * 256 + e) * NP + p];
        int m = m0 + rr;
        out[(size_t)(m * NF + s) * NP + p] = a3;
    }
}

// ---------------------------------------------------------------------------
// Reward MLP: 448 -> 256 -> 256 -> 1, lrelu.  One block per batch element.
// ---------------------------------------------------------------------------
__global__ void reward_kernel(const float* __restrict__ r32,      // (B,7,64)
                              const float* __restrict__ W1, const float* __restrict__ b1,
                              const float* __restrict__ W2, const float* __restrict__ b2,
                              const float* __restrict__ W3, const float* __restrict__ b3,
                              float* __restrict__ out) {
    __shared__ float sg[DGE];
    __shared__ float sh[DH];
    __shared__ float sh2[DH];
    int b = blockIdx.x, t = threadIdx.x;
    for (int i = t; i < DGE; i += 256) sg[i] = r32[(size_t)b * DGE + i];
    __syncthreads();
    float a = b1[t];
    for (int i = 0; i < DGE; ++i) a += sg[i] * W1[i * DH + t];
    sh[t] = a >= 0.f ? a : 0.01f * a;
    __syncthreads();
    float c = b2[t];
    for (int i = 0; i < DH; ++i) c += sh[i] * W2[i * DH + t];
    c = c >= 0.f ? c : 0.01f * c;
    sh2[t] = c * W3[t];
    __syncthreads();
    for (int st = 128; st > 0; st >>= 1) {
        if (t < st) sh2[t] += sh2[t + st];
        __syncthreads();
    }
    if (t == 0) out[b] = sh2[0] + b3[0];
}

// ---------------------------------------------------------------------------
// Host launcher
// ---------------------------------------------------------------------------
extern "C" void kernel_launch(void* const* d_in, const int* in_sizes, int n_in,
                              void* d_out, int out_size, void* d_ws, size_t ws_size,
                              hipStream_t stream) {
    (void)in_sizes; (void)n_in; (void)out_size; (void)ws_size;

    const float* x_A      = (const float*)d_in[0];
    const float* x_B      = (const float*)d_in[1];
    const int*   gmask    = (const int*)d_in[2];
    const int*   lmask_A  = (const int*)d_in[3];
    const int*   lmask_B  = (const int*)d_in[4];
    const int*   objA     = (const int*)d_in[5];
    const int*   objB     = (const int*)d_in[6];
    const float* enc_Wl_A = (const float*)d_in[7];
    const float* enc_bl_A = (const float*)d_in[8];
    const float* enc_Wg_A = (const float*)d_in[9];
    const float* enc_bg_A = (const float*)d_in[10];
    const float* enc_Wl_B = (const float*)d_in[11];
    const float* enc_bl_B = (const float*)d_in[12];
    const float* enc_Wg_B = (const float*)d_in[13];
    const float* enc_bg_B = (const float*)d_in[14];
    const float* dec_W1_A = (const float*)d_in[15];
    const float* dec_b1_A = (const float*)d_in[16];
    const float* dec_W2_A = (const float*)d_in[17];
    const float* dec_b2_A = (const float*)d_in[18];
    const float* dec_Wp_A = (const float*)d_in[19];
    const float* dec_bp_A = (const float*)d_in[20];
    const float* dec_W1_B = (const float*)d_in[21];
    const float* dec_b1_B = (const float*)d_in[22];
    const float* dec_W2_B = (const float*)d_in[23];
    const float* dec_b2_B = (const float*)d_in[24];
    const float* dec_Wp_B = (const float*)d_in[25];
    const float* dec_bp_B = (const float*)d_in[26];
    const float* rw_W1    = (const float*)d_in[27];
    const float* rw_b1    = (const float*)d_in[28];
    const float* rw_W2    = (const float*)d_in[29];
    const float* rw_b2    = (const float*)d_in[30];
    const float* rw_W3    = (const float*)d_in[31];
    const float* rw_b3    = (const float*)d_in[32];

    // workspace carve-up (256-B aligned)
    char*  ws  = (char*)d_ws;
    size_t off = 0;
    auto carve = [&](size_t bytes) -> void* {
        void* p = ws + off;
        off += (bytes + 255) & ~(size_t)255;
        return p;
    };
    __bf16* WencA = (__bf16*)carve(128 * 128 * 2);
    __bf16* WencB = (__bf16*)carve(128 * 128 * 2);
    __bf16* W1tA  = (__bf16*)carve((size_t)NF_A * DH * DINdA * 2);
    __bf16* W2tA  = (__bf16*)carve((size_t)NF_A * DH * DH * 2);
    __bf16* W1tB  = (__bf16*)carve((size_t)NF_B * DH * DINdB * 2);
    __bf16* W2tB  = (__bf16*)carve((size_t)NF_B * DH * DH * 2);
    __bf16* hlA   = (__bf16*)carve((size_t)MA * 64 * 2);
    float*  hgA   = (float*) carve((size_t)MA * 64 * 4);
    __bf16* hlB   = (__bf16*)carve((size_t)MB * 64 * 2);
    float*  hgB   = (float*) carve((size_t)MB * 64 * 4);
    float*  r32   = (float*) carve((size_t)Bn * NS * 64 * 4);
    __bf16* rbf   = (__bf16*)carve((size_t)Bn * NS * 64 * 2);

    // ---- weight prep (bf16, N-major)
    pack_encw_kernel<<<64, 256, 0, stream>>>(enc_Wl_A, enc_Wg_A, WencA);
    pack_encw_kernel<<<64, 256, 0, stream>>>(enc_Wl_B, enc_Wg_B, WencB);
    transpose_bf16_kernel<<<512, 256, 0, stream>>>(dec_W1_A, W1tA, NF_A, DINdA, DH);
    transpose_bf16_kernel<<<256, 256, 0, stream>>>(dec_W2_A, W2tA, NF_A, DH, DH);
    transpose_bf16_kernel<<<512, 256, 0, stream>>>(dec_W1_B, W1tB, NF_B, DINdB, DH);
    transpose_bf16_kernel<<<256, 256, 0, stream>>>(dec_W2_B, W2tB, NF_B, DH, DH);

    // ---- encoders (WMMA, LDS-staged weights)
    enc_gemm_kernel<<<MA / 64, 256, 0, stream>>>(x_A, WencA, enc_bl_A, enc_bg_A, hlA, hgA);
    enc_gemm_kernel<<<MB / 64, 256, 0, stream>>>(x_B, WencB, enc_bl_B, enc_bg_B, hlB, hgB);

    // ---- global pooling
    reduce_global_kernel<<<dim3(Bn, NS), 64, 0, stream>>>(hgA, hgB, objA, objB, r32, rbf);

    // ---- decoders (fused 3-layer WMMA MLPs)
    float* outA = (float*)d_out;
    float* outB = outA + (size_t)Bn * NO_A * NF_A * NP;
    float* outR = outB + (size_t)Bn * NO_B * NF_B * NP;

    decoder_kernel<<<dim3(Bn * NO_A / 32, NF_A), 256, 0, stream>>>(
        hlA, rbf, lmask_A, gmask, W1tA, dec_b1_A, W2tA, dec_b2_A, dec_Wp_A, dec_bp_A,
        outA, NO_A, NF_A, DINdA, 0);
    decoder_kernel<<<dim3(Bn * NO_B / 32, NF_B), 256, 0, stream>>>(
        hlB, rbf, lmask_B, gmask, W1tB, dec_b1_B, W2tB, dec_b2_B, dec_Wp_B, dec_bp_B,
        outB, NO_B, NF_B, DINdB, NF_A);

    // ---- reward head
    reward_kernel<<<Bn, 256, 0, stream>>>(r32, rw_W1, rw_b1, rw_W2, rw_b2, rw_W3, rw_b3, outR);
}
